// temporal_self_attention_6811818131526
// MI455X (gfx1250) — matrix operand order repaired and checked by reading, not compiled, not run on hardware
//
#include <hip/hip_runtime.h>

#define D_MODEL 1024
#define SEQ_T   2048
#define NBATCH  2
#define NHEAD   16
#define HEAD_D  64
#define TTOT    (NBATCH*SEQ_T)

typedef _Float16 v16h __attribute__((ext_vector_type(16)));
typedef _Float16 v8h  __attribute__((ext_vector_type(8)));
typedef _Float16 v4h  __attribute__((ext_vector_type(4)));
typedef float    v8f  __attribute__((ext_vector_type(8)));
typedef float    v4f  __attribute__((ext_vector_type(4)));
typedef unsigned int v4u __attribute__((ext_vector_type(4)));
typedef int      v8i  __attribute__((ext_vector_type(8)));
typedef int      v4i  __attribute__((ext_vector_type(4)));

union V16 { v16h v; v8h h[2]; };

#define WMMA_F16(A, B, C) \
  __builtin_amdgcn_wmma_f32_16x16x32_f16(false, (A), false, (B), (short)0, (C), false, false)

// ---------------------------------------------------------------------------
// TDM: issue a 2-D tensor tile load (16-bit elements) global -> LDS.
// D# packed per CDNA5 ISA 8.3/8.4 (group0: count/lds/global/type,
// group1: data_size, tensor dims, tile dims, row stride). Groups 2/3 zero
// (2-D tensor). Tracked by TENSORcnt. This toolchain uses the 6-arg builtin
// (g0, g1, g2, g3, extra group, cpol).
// ---------------------------------------------------------------------------
__device__ __forceinline__ void tdm_load_2d_f16(const void* gaddr, unsigned lds_off,
                                                unsigned tensor_d0, unsigned tensor_d1,
                                                unsigned tile_d0, unsigned tile_d1,
                                                unsigned row_stride_elems) {
  const unsigned long long ga = (unsigned long long)gaddr;
  v4u g0;
  g0[0] = 1u;                                      // count=1, no gather
  g0[1] = lds_off;                                 // lds_addr (bytes)
  g0[2] = (unsigned)ga;                            // global_addr[31:0]
  g0[3] = (unsigned)(ga >> 32) | 0x80000000u;      // global_addr[56:32] | type=2
  v8i g1;
  g1[0] = 0x00010000;                              // data_size = 1 (2 bytes)
  g1[1] = (int)((tensor_d0 & 0xFFFFu) << 16);      // tensor_dim0[15:0]
  g1[2] = (int)(((tensor_d0 >> 16) & 0xFFFFu) | ((tensor_d1 & 0xFFFFu) << 16));
  g1[3] = (int)(((tensor_d1 >> 16) & 0xFFFFu) | ((tile_d0 & 0xFFFFu) << 16));
  g1[4] = (int)(tile_d1 & 0xFFFFu);                // tile_dim1 (tile_dim2 = 0)
  g1[5] = (int)row_stride_elems;                   // tensor_dim0_stride[31:0]
  g1[6] = 0;                                       // stride hi / dim1_stride lo
  g1[7] = 0;
  v4i z4 = {0, 0, 0, 0};
  v8i z8 = {0, 0, 0, 0, 0, 0, 0, 0};
  __builtin_amdgcn_tensor_load_to_lds(g0, g1, z4, z4, z8, 0);
}

// ---------------------------------------------------------------------------
// LayerNorm (fp32 in) -> fp16 normalized activations. One block per row.
// ---------------------------------------------------------------------------
__global__ void ln_to_f16(const float* __restrict__ x,
                          const float* __restrict__ gamma,
                          const float* __restrict__ beta,
                          _Float16* __restrict__ xn) {
  const int row = blockIdx.x;
  const int tid = threadIdx.x;                       // 256 threads, 4 cols each
  const float* xr = x + (size_t)row * D_MODEL;
  v4f xv = *(const v4f*)(xr + tid * 4);
  float s  = xv[0] + xv[1] + xv[2] + xv[3];
  float s2 = xv[0]*xv[0] + xv[1]*xv[1] + xv[2]*xv[2] + xv[3]*xv[3];
  #pragma unroll
  for (int m = 1; m < 32; m <<= 1) {
    s  += __shfl_xor(s,  m, 32);
    s2 += __shfl_xor(s2, m, 32);
  }
  __shared__ float red[16];
  if ((tid & 31) == 0) { red[tid >> 5] = s; red[8 + (tid >> 5)] = s2; }
  __syncthreads();
  float ts = 0.f, ts2 = 0.f;
  #pragma unroll
  for (int i = 0; i < 8; ++i) { ts += red[i]; ts2 += red[8 + i]; }
  const float mu  = ts  * (1.0f / D_MODEL);
  const float var = ts2 * (1.0f / D_MODEL) - mu * mu;
  const float rs  = rsqrtf(var + 1e-5f);
  v4h o;
  #pragma unroll
  for (int j = 0; j < 4; ++j) {
    const int c = tid * 4 + j;
    o[j] = (_Float16)((xv[j] - mu) * rs * gamma[c] + beta[c]);
  }
  *(v4h*)(xn + (size_t)row * D_MODEL + tid * 4) = o;
}

// ---------------------------------------------------------------------------
// fp32 -> fp16 conversion (weights)
// ---------------------------------------------------------------------------
__global__ void cvt_f16(const float* __restrict__ src, _Float16* __restrict__ dst, int n4) {
  const int i = blockIdx.x * blockDim.x + threadIdx.x;
  if (i < n4) {
    v4f s = *(const v4f*)(src + (size_t)i * 4);
    v4h d;
    #pragma unroll
    for (int j = 0; j < 4; ++j) d[j] = (_Float16)s[j];
    *(v4h*)(dst + (size_t)i * 4) = d;
  }
}

// ---------------------------------------------------------------------------
// C = A (M x K, row-major f16) * W^T (W is N x K row-major f16).
// Block = 4 waves, each wave computes 32x64 (2 M-tiles x 4 N-tiles = 8 WMMA
// per 32-wide K step). The 64-row W panel shared by all 4 waves is staged in
// LDS by the Tensor Data Mover in 64x256 chunks, ping-pong double-buffered
// and overlapped with WMMA compute (TENSORcnt + barrier handoff). A stays in
// global (L2-resident). transOut: 0 = row-major out (Q,K); 1 = feature-major
// transposed out (V).
// ---------------------------------------------------------------------------
#define KCH 256   // K chunk staged per TDM transfer

__global__ void qkv_gemm(const _Float16* __restrict__ A,
                         const _Float16* __restrict__ W,
                         _Float16* __restrict__ Crm,
                         _Float16* __restrict__ Ct,
                         int transOut) {
  const int w    = threadIdx.x >> 5;
  const int lane = threadIdx.x & 31;
  const int hf   = lane >> 4;     // 16-lane half
  const int l15  = lane & 15;
  const int m0   = blockIdx.x * 128 + w * 32;   // two 16-row tiles per wave
  const int n0   = blockIdx.y * 64;             // four 16-col tiles per wave

  __shared__ _Float16 wpanel[2][64 * KCH];      // 2 x 32 KB ping-pong

  // A fragment rows (lane = M); ISA 16-bit A layout: regs0-3 = K hf*8..+7,
  // regs4-7 = K 16+hf*8..+7 within each 32-wide K step.
  const _Float16* ar0 = A + (size_t)(m0 + l15) * D_MODEL;
  const _Float16* ar1 = ar0 + (size_t)16 * D_MODEL;
  const _Float16* wbase = W + (size_t)n0 * D_MODEL;

  // Prime the pipeline: chunk 0 into panel 0.
  if (w == 0) {
    tdm_load_2d_f16(wbase,
                    (unsigned)(size_t)(&wpanel[0][0]),
                    KCH, 64, KCH, 64, D_MODEL);
    __builtin_amdgcn_s_wait_tensorcnt(0);
  }
  __syncthreads();

  v8f acc[2][4] = {};
  const int NCH = D_MODEL / KCH;
  for (int c = 0; c < NCH; ++c) {
    // Kick off the next chunk while computing on the current one.
    if (c + 1 < NCH && w == 0)
      tdm_load_2d_f16(wbase + (size_t)(c + 1) * KCH,
                      (unsigned)(size_t)(&wpanel[(c + 1) & 1][0]),
                      KCH, 64, KCH, 64, D_MODEL);

    const _Float16* wp = &wpanel[c & 1][0];
    const int kb = c * KCH;
    #pragma unroll 2
    for (int k = 0; k < KCH; k += 32) {
      V16 a0, a1;
      a0.h[0] = *(const v8h*)(ar0 + kb + k + hf * 8);
      a0.h[1] = *(const v8h*)(ar0 + kb + k + hf * 8 + 16);
      a1.h[0] = *(const v8h*)(ar1 + kb + k + hf * 8);
      a1.h[1] = *(const v8h*)(ar1 + kb + k + hf * 8 + 16);
      #pragma unroll
      for (int nt = 0; nt < 4; ++nt) {
        // B fragment from LDS: lane = N row (nt*16 + l15), K hf*8..+7 and 16+hf*8..+7.
        const _Float16* wrow = wp + (size_t)(nt * 16 + l15) * KCH + k;
        V16 b;
        b.h[0] = *(const v8h*)(wrow + hf * 8);
        b.h[1] = *(const v8h*)(wrow + hf * 8 + 16);
        acc[0][nt] = WMMA_F16(a0.v, b.v, acc[0][nt]);
        acc[1][nt] = WMMA_F16(a1.v, b.v, acc[1][nt]);
      }
    }

    if (c + 1 < NCH) {
      if (w == 0) __builtin_amdgcn_s_wait_tensorcnt(0);
      __syncthreads();   // publish next panel; all waves done with old one
    }
  }

  // C layout: reg r -> M = hf*8 + r, N = l15
  if (!transOut) {
    #pragma unroll
    for (int mt = 0; mt < 2; ++mt)
      #pragma unroll
      for (int nt = 0; nt < 4; ++nt)
        #pragma unroll
        for (int r = 0; r < 8; ++r)
          Crm[(size_t)(m0 + mt * 16 + hf * 8 + r) * D_MODEL + n0 + nt * 16 + l15] =
              (_Float16)acc[mt][nt][r];
  } else {
    #pragma unroll
    for (int nt = 0; nt < 4; ++nt) {
      _Float16* col = Ct + (size_t)(n0 + nt * 16 + l15) * TTOT + m0 + hf * 8;
      #pragma unroll
      for (int mt = 0; mt < 2; ++mt)
        #pragma unroll
        for (int r = 0; r < 8; ++r)
          col[mt * 16 + r] = (_Float16)acc[mt][nt][r];
    }
  }
}

// ---------------------------------------------------------------------------
// Flash attention: one wave per (batch, head, 32-query block) -> every K/V
// fragment feeds two query tiles (1 b128 load per WMMA). Per 32-key step:
// 8 WMMA for S = Q*K^T, online softmax, P via per-wave LDS transpose
// (C-layout -> A-layout), 8 WMMA for O += P*V. V is feature-major (Vt) so
// the PV B-fragment is contiguous along keys. 4 waves per 128-thread block.
// ---------------------------------------------------------------------------
__global__ void attn_fwd(const _Float16* __restrict__ Qm,
                         const _Float16* __restrict__ Km,
                         const _Float16* __restrict__ Vt,
                         const float* __restrict__ x,
                         float* __restrict__ out) {
  const int w    = threadIdx.x >> 5;
  const int lane = threadIdx.x & 31;
  const int hf   = lane >> 4;
  const int l15  = lane & 15;
  const int h    = blockIdx.y;
  const int b    = blockIdx.z;
  const int tok0 = b * SEQ_T + (blockIdx.x * 4 + w) * 32;   // 32 queries/wave

  __shared__ _Float16 pbuf_all[4][32 * 32];                 // 2 KB per wave
  _Float16* pbuf = pbuf_all[w];

  // Q A-fragments: [query tile][32-wide K step over head_dim=64]
  V16 qa[2][2];
  #pragma unroll
  for (int qt = 0; qt < 2; ++qt) {
    const _Float16* qrow = Qm + (size_t)(tok0 + qt * 16 + l15) * D_MODEL + h * HEAD_D;
    #pragma unroll
    for (int s = 0; s < 2; ++s) {
      qa[qt][s].h[0] = *(const v8h*)(qrow + s * 32 + hf * 8);
      qa[qt][s].h[1] = *(const v8h*)(qrow + s * 32 + hf * 8 + 16);
    }
  }

  v8f o[2][4] = {};
  float mrow[2][8], lrow[2][8];
  #pragma unroll
  for (int qt = 0; qt < 2; ++qt)
    #pragma unroll
    for (int r = 0; r < 8; ++r) { mrow[qt][r] = -1e30f; lrow[qt][r] = 0.f; }

  const float scale = 0.125f;   // 1/sqrt(64)

  for (int kt = 0; kt < SEQ_T; kt += 32) {
    // ---- scores: [query tile][key tile], each 16x16 ----
    v8f sAcc[2][2] = {};
    #pragma unroll
    for (int e = 0; e < 2; ++e) {
      const _Float16* krow =
          Km + (size_t)(b * SEQ_T + kt + e * 16 + l15) * D_MODEL + h * HEAD_D;
      #pragma unroll
      for (int s = 0; s < 2; ++s) {
        V16 kb;
        kb.h[0] = *(const v8h*)(krow + s * 32 + hf * 8);
        kb.h[1] = *(const v8h*)(krow + s * 32 + hf * 8 + 16);
        sAcc[0][e] = WMMA_F16(qa[0][s].v, kb.v, sAcc[0][e]);
        sAcc[1][e] = WMMA_F16(qa[1][s].v, kb.v, sAcc[1][e]);
      }
    }
    __builtin_amdgcn_wave_barrier();
    // ---- online softmax; row M = hf*8 + r, all 16 N values in one half ----
    #pragma unroll
    for (int qt = 0; qt < 2; ++qt) {
      #pragma unroll
      for (int r = 0; r < 8; ++r) {
        const float a = sAcc[qt][0][r] * scale;
        const float c = sAcc[qt][1][r] * scale;
        float t = fmaxf(a, c);
        t = fmaxf(t, __shfl_xor(t, 1, 32));
        t = fmaxf(t, __shfl_xor(t, 2, 32));
        t = fmaxf(t, __shfl_xor(t, 4, 32));
        t = fmaxf(t, __shfl_xor(t, 8, 32));
        const float nm    = fmaxf(mrow[qt][r], t);
        const float alpha = __expf(mrow[qt][r] - nm);
        mrow[qt][r] = nm;
        const float p0 = __expf(a - nm);
        const float p1 = __expf(c - nm);
        float ps = p0 + p1;
        ps += __shfl_xor(ps, 1, 32);
        ps += __shfl_xor(ps, 2, 32);
        ps += __shfl_xor(ps, 4, 32);
        ps += __shfl_xor(ps, 8, 32);
        lrow[qt][r] = lrow[qt][r] * alpha + ps;
        o[qt][0][r] *= alpha; o[qt][1][r] *= alpha;
        o[qt][2][r] *= alpha; o[qt][3][r] *= alpha;
        pbuf[(qt * 16 + hf * 8 + r) * 32 + l15]      = (_Float16)p0;
        pbuf[(qt * 16 + hf * 8 + r) * 32 + 16 + l15] = (_Float16)p1;
      }
    }
    asm volatile("s_wait_dscnt 0x0" ::: "memory");   // LDS RAW fence (per-wave)
    __builtin_amdgcn_wave_barrier();
    // ---- P (16x32 per query tile) A-fragments from LDS (transposed) ----
    V16 pa0, pa1;
    pa0.h[0] = *(const v8h*)&pbuf[l15 * 32 + hf * 8];
    pa0.h[1] = *(const v8h*)&pbuf[l15 * 32 + hf * 8 + 16];
    pa1.h[0] = *(const v8h*)&pbuf[(16 + l15) * 32 + hf * 8];
    pa1.h[1] = *(const v8h*)&pbuf[(16 + l15) * 32 + hf * 8 + 16];
    // ---- O += P*V : 4 head-dim column tiles; Vt contiguous along keys ----
    const _Float16* vbase = Vt + (size_t)(h * HEAD_D + l15) * TTOT
                               + b * SEQ_T + kt + hf * 8;
    #pragma unroll
    for (int nt = 0; nt < 4; ++nt) {
      const _Float16* vp = vbase + (size_t)(nt * 16) * TTOT;
      V16 vb;
      vb.h[0] = *(const v8h*)(vp);
      vb.h[1] = *(const v8h*)(vp + 16);
      o[0][nt] = WMMA_F16(pa0.v, vb.v, o[0][nt]);
      o[1][nt] = WMMA_F16(pa1.v, vb.v, o[1][nt]);
    }
    __builtin_amdgcn_wave_barrier();   // WAR: pbuf rewritten next iteration
  }

  // ---- normalize, residual add, fp32 store ----
  #pragma unroll
  for (int qt = 0; qt < 2; ++qt) {
    #pragma unroll
    for (int r = 0; r < 8; ++r) {
      const int t = tok0 + qt * 16 + hf * 8 + r;
      const float inv = 1.0f / lrow[qt][r];
      float* orow = out + (size_t)t * D_MODEL + h * HEAD_D;
      const float* xrow = x + (size_t)t * D_MODEL + h * HEAD_D;
      #pragma unroll
      for (int nt = 0; nt < 4; ++nt)
        orow[nt * 16 + l15] = xrow[nt * 16 + l15] + o[qt][nt][r] * inv;
    }
  }
}

// ---------------------------------------------------------------------------
extern "C" void kernel_launch(void* const* d_in, const int* in_sizes, int n_in,
                              void* d_out, int out_size, void* d_ws, size_t ws_size,
                              hipStream_t stream) {
  (void)in_sizes; (void)n_in; (void)out_size; (void)ws_size;
  const float* x     = (const float*)d_in[0];
  const float* Wq    = (const float*)d_in[1];
  const float* Wk    = (const float*)d_in[2];
  const float* Wv    = (const float*)d_in[3];
  const float* gamma = (const float*)d_in[4];
  const float* beta  = (const float*)d_in[5];
  float* out = (float*)d_out;

  char* ws = (char*)d_ws;
  size_t off = 0;
  _Float16* xn  = (_Float16*)(ws + off); off += (size_t)TTOT * D_MODEL * 2;     // 8 MB
  _Float16* wqh = (_Float16*)(ws + off); off += (size_t)D_MODEL * D_MODEL * 2;  // 2 MB
  _Float16* wkh = (_Float16*)(ws + off); off += (size_t)D_MODEL * D_MODEL * 2;
  _Float16* wvh = (_Float16*)(ws + off); off += (size_t)D_MODEL * D_MODEL * 2;
  _Float16* qh  = (_Float16*)(ws + off); off += (size_t)TTOT * D_MODEL * 2;     // 8 MB
  _Float16* kh  = (_Float16*)(ws + off); off += (size_t)TTOT * D_MODEL * 2;     // 8 MB
  _Float16* vth = (_Float16*)(ws + off); off += (size_t)TTOT * D_MODEL * 2;     // 8 MB

  // 1) LayerNorm -> fp16
  ln_to_f16<<<TTOT, 256, 0, stream>>>(x, gamma, beta, xn);

  // 2) weights fp32 -> fp16
  const int n4 = D_MODEL * D_MODEL / 4;
  cvt_f16<<<(n4 + 255) / 256, 256, 0, stream>>>(Wq, wqh, n4);
  cvt_f16<<<(n4 + 255) / 256, 256, 0, stream>>>(Wk, wkh, n4);
  cvt_f16<<<(n4 + 255) / 256, 256, 0, stream>>>(Wv, wvh, n4);

  // 3) QKV projections (WMMA + TDM-staged W panels). V stored feature-major.
  dim3 ggrid(TTOT / 128, D_MODEL / 64);
  qkv_gemm<<<ggrid, 128, 0, stream>>>(xn, wqh, qh, nullptr, 0);
  qkv_gemm<<<ggrid, 128, 0, stream>>>(xn, wkh, kh, nullptr, 0);
  qkv_gemm<<<ggrid, 128, 0, stream>>>(xn, wvh, nullptr, vth, 1);

  // 4) flash attention + residual (32 queries per wave, 4 waves per block)
  dim3 agrid(SEQ_T / 128, NHEAD, NBATCH);
  attn_fwd<<<agrid, 128, 0, stream>>>(qh, kh, vth, x, out);
}
